// YOLOv2Loss_33346126086533
// MI455X (gfx1250) — compile-verified, matrix operand order checked by reading, not run on hardware
//
#include <hip/hip_runtime.h>
#include <math.h>

#define BATCH   256
#define NA      5
#define NC      80
#define SG      13
#define SS      (SG*SG)        /* 169 */
#define CHN     (5+NC)         /* 85 */
#define NCELLS  (BATCH*NA*SS)  /* 216320 */
#define L_COORD 5.0f
#define L_NOOBJ 0.5f

typedef __attribute__((ext_vector_type(16))) _Float16 v16h;
typedef __attribute__((ext_vector_type(8)))  float    v8f;

__device__ __forceinline__ float sigmoidf_(float x) { return 1.0f / (1.0f + __expf(-x)); }

// ---------------------------------------------------------------- init
__global__ void k_init(float* mask, int* counter, float* out) {
    int i = blockIdx.x * blockDim.x + threadIdx.x;
    if (i < NCELLS) mask[i] = 0.0f;
    if (i == 0) { *counter = 0; *out = 0.0f; }
}

// ------------------------------------------------------- target scatter
__global__ void k_build_targets(const float* __restrict__ tgt,
                                const float* __restrict__ anchors,
                                float* mask, float* tx, float* ty,
                                float* tw, float* th, int* tcls,
                                int* list, int* counter, int ngt) {
    int i = blockIdx.x * blockDim.x + threadIdx.x;
    if (i >= ngt) return;
    const float Sf = (float)SG;
    int   b   = (int)tgt[i*6 + 0];
    int   cls = (int)tgt[i*6 + 1];
    float gx = tgt[i*6 + 2] * Sf, gy = tgt[i*6 + 3] * Sf;
    float gw = tgt[i*6 + 4] * Sf, gh = tgt[i*6 + 5] * Sf;
    int gi = min(max((int)gx, 0), SG - 1);
    int gj = min(max((int)gy, 0), SG - 1);
    float best_iou = -1.0f; int best = 0;
#pragma unroll
    for (int a = 0; a < NA; a++) {
        float aw = anchors[2*a] * Sf, ah = anchors[2*a + 1] * Sf;
        float inter = fminf(gw, aw) * fminf(gh, ah);
        float uni   = gw * gh + aw * ah - inter;
        float iou   = inter / (uni + 1e-6f);
        if (iou > best_iou) { best_iou = iou; best = a; }
    }
    float aw = anchors[2*best] * Sf, ah = anchors[2*best + 1] * Sf;
    int flat = ((b * NA + best) * SG + gj) * SG + gi;
    tx[flat] = gx - (float)gi;
    ty[flat] = gy - (float)gj;
    tw[flat] = __logf(gw / aw + 1e-16f);
    th[flat] = __logf(gh / ah + 1e-16f);
    tcls[flat] = cls;
    float old = atomicExch(&mask[flat], 1.0f);
    if (old == 0.0f) {                 // unique appender -> dedup'd compact list
        int p = atomicAdd(counter, 1);
        list[p] = flat;
    }
}

// -------------------------------------- noobj conf loss over ALL cells
// Reads only channel 4 of 85 -> 0.87 MB instead of 73.5 MB.
__global__ void k_conf_noobj(const float* __restrict__ pred, float* out) {
    int idx = blockIdx.x * blockDim.x + threadIdx.x;
    float v = 0.0f;
    if (idx < NCELLS) {
        int ba = idx / SS;
        int s  = idx - ba * SS;
        float c = sigmoidf_(pred[(ba * CHN + 4) * SS + s]);
        v = L_NOOBJ * c * c;
    }
    __shared__ float sm[256];
    sm[threadIdx.x] = v;
    __syncthreads();
    for (int o = 128; o > 0; o >>= 1) {
        if (threadIdx.x < o) sm[threadIdx.x] += sm[threadIdx.x + o];
        __syncthreads();
    }
    if (threadIdx.x == 0) atomicAdd(out, sm[0] * (1.0f / (float)BATCH));
}

// ----------------------- masked-cell loss: coord + conf-corr + softmax
// One wave handles 16 masked cells. The 80-class (pad->96) exp-sum for all
// 16 cells is computed as D = A x ones via 3x v_wmma_f32_16x16x32_f16.
__global__ void k_masked_loss(const float* __restrict__ pred,
                              const float* __restrict__ tx, const float* __restrict__ ty,
                              const float* __restrict__ tw, const float* __restrict__ th,
                              const int*   __restrict__ tcls,
                              const int*   __restrict__ list,
                              const int*   __restrict__ counter,
                              float* out) {
    const int lane  = threadIdx.x & 31;
    const int wave  = threadIdx.x >> 5;
    const int group = blockIdx.x * (blockDim.x >> 5) + wave;
    const int M     = *counter;
    const int base  = group * 16;
    if (base >= M) return;                 // wave-uniform -> EXEC stays all-1s

    const int  m     = lane & 15;          // row (cell within group)
    const int  h     = lane >> 4;          // which K-half this lane supplies
    const int  cell  = base + m;
    const bool valid = (cell < M);
    const int  f     = list[valid ? cell : base];
    const int  ba    = f / SS;
    const int  s     = f - ba * SS;
    const float* cls0 = pred + (ba * CHN + 5) * SS + s;   // class c at cls0[c*SS]

    // ---- load this lane's 48 class slots (3 chunks x 16), track max ----
    float vals[48];
    float lmax = -3.0e38f;
#pragma unroll
    for (int k = 0; k < 3; k++) {
#pragma unroll
        for (int e = 0; e < 16; e++) {
            int K = e + ((e < 8) ? (h ? 8 : 0) : (h ? 16 : 8));   // ISA 16-bit A layout
            int c = 32 * k + K;
            float x = (c < NC && valid) ? cls0[c * SS] : -3.0e38f;
            vals[k * 16 + e] = x;
            lmax = fmaxf(lmax, x);
        }
    }
    float cmax = fmaxf(lmax, __shfl_xor(lmax, 16, 32));   // cell max (lane pair)

    // ---- x[tcls] for this cell (one of the pair holds it) ----
    int tc = tcls[f];
    float xtc = 0.0f;
#pragma unroll
    for (int k = 0; k < 3; k++) {
#pragma unroll
        for (int e = 0; e < 16; e++) {
            int K = e + ((e < 8) ? (h ? 8 : 0) : (h ? 16 : 8));
            int c = 32 * k + K;
            if (c == tc) xtc = vals[k * 16 + e];
        }
    }
    xtc += __shfl_xor(xtc, 16, 32);

    // ---- batched exp-sum: D = exp(A - max) x ones, f32 accumulate ----
    v16h bones;
#pragma unroll
    for (int e = 0; e < 16; e++) bones[e] = (_Float16)1.0f;
    v8f d = {};
#pragma unroll
    for (int k = 0; k < 3; k++) {
        v16h a;
#pragma unroll
        for (int e = 0; e < 16; e++) {
            int K = e + ((e < 8) ? (h ? 8 : 0) : (h ? 16 : 8));
            int c = 32 * k + K;
            float ev = (c < NC && valid) ? __expf(vals[k * 16 + e] - cmax) : 0.0f;
            a[e] = (_Float16)ev;
        }
        d = __builtin_amdgcn_wmma_f32_16x16x32_f16(
                /*neg_a=*/false, a, /*neg_b=*/false, bones,
                /*c_mod=*/(short)0, d, /*reuse_a=*/false, /*reuse_b=*/false);
    }

    // ---- sum of log(sumexp) over the 16 cells ----
    // lanes 0-15: d[v] = sumexp of cell base+v ; lanes 16-31: cell base+8+v
    float llog = 0.0f;
#pragma unroll
    for (int v = 0; v < 8; v++) {
        int cc = base + (h ? 8 : 0) + v;
        llog += (cc < M) ? __logf(d[v]) : 0.0f;
    }
    float lg = __shfl(llog, 0, 32) + __shfl(llog, 16, 32);

    // ---- per-cell coord + conf-correction + (max - x_tc), lanes h==0 ----
    const float* ch = pred + ba * CHN * SS + s;
    float p0 = ch[0], p1 = ch[SS], p2 = ch[2 * SS], p3 = ch[3 * SS], p4 = ch[4 * SS];
    float sx = sigmoidf_(p0), sy = sigmoidf_(p1), cf = sigmoidf_(p4);
    float dx = sx - tx[f], dy = sy - ty[f], dw = p2 - tw[f], dh = p3 - th[f];
    float coord = L_COORD * (dx*dx + dy*dy + dw*dw + dh*dh);
    float confc = (cf - 1.0f) * (cf - 1.0f) - L_NOOBJ * cf * cf;  // noobj pass already added L_NOOBJ*cf^2
    float clsc  = cmax - xtc;                                     // -logp = max + log(sumexp) - x_tc
    float acc   = (h == 0 && valid) ? (coord + confc + clsc) : 0.0f;

#pragma unroll
    for (int o = 16; o > 0; o >>= 1) acc += __shfl_xor(acc, o, 32);
    if (lane == 0) atomicAdd(out, (acc + lg) * (1.0f / (float)BATCH));
}

// ---------------------------------------------------------------- launch
extern "C" void kernel_launch(void* const* d_in, const int* in_sizes, int n_in,
                              void* d_out, int out_size, void* d_ws, size_t ws_size,
                              hipStream_t stream) {
    const float* pred    = (const float*)d_in[0];
    const float* target  = (const float*)d_in[1];
    const float* anchors = (const float*)d_in[2];
    (void)d_in; (void)n_in; (void)out_size; (void)ws_size;
    const int ngt = in_sizes[1] / 6;

    // workspace layout (floats/ints), ~5.3 MB
    float* mask = (float*)d_ws;
    float* tx   = mask + NCELLS;
    float* ty   = tx   + NCELLS;
    float* tw   = ty   + NCELLS;
    float* th   = tw   + NCELLS;
    int*   tcls = (int*)(th + NCELLS);
    int*   list = tcls + NCELLS;
    int*   counter = list + ngt;
    float* out = (float*)d_out;

    const int T = 256;
    k_init<<<(NCELLS + T - 1) / T, T, 0, stream>>>(mask, counter, out);
    k_build_targets<<<(ngt + T - 1) / T, T, 0, stream>>>(
        target, anchors, mask, tx, ty, tw, th, tcls, list, counter, ngt);
    k_conf_noobj<<<(NCELLS + T - 1) / T, T, 0, stream>>>(pred, out);
    const int groups = (ngt + 15) / 16;       // upper bound on masked groups
    const int blocks = (groups + 7) / 8;      // 8 waves per 256-thread block
    k_masked_loss<<<blocks, T, 0, stream>>>(pred, tx, ty, tw, th, tcls, list, counter, out);
}
